// GAT_35184372089500
// MI455X (gfx1250) — matrix operand order repaired and checked by reading, not compiled, not run on hardware
//
#include <hip/hip_runtime.h>

#define NN 20000
#define NE 240000
#define FIN 128
#define EMB 256
#define NH 2
#define NG 128
#define DENSE 512
#define NCLS 10
#define ETOT (NE + NN)
#define HE (NH * EMB)   // 512

typedef __attribute__((ext_vector_type(16))) __bf16 v16bf;
typedef __attribute__((ext_vector_type(8)))  float  v8f;

// ---------- helpers ----------
__device__ __forceinline__ unsigned short f2bf(float f) {
  unsigned u = __float_as_uint(f);
  u += 0x7FFFu + ((u >> 16) & 1u);          // round-to-nearest-even
  return (unsigned short)(u >> 16);
}
// order-preserving float<->uint encoding for atomicMax on floats
__device__ __forceinline__ unsigned fenc(float f) {
  unsigned u = __float_as_uint(f);
  return (u & 0x80000000u) ? ~u : (u | 0x80000000u);
}
__device__ __forceinline__ float fdec(unsigned u) {
  unsigned v = (u & 0x80000000u) ? (u & 0x7FFFFFFFu) : ~u;
  return __uint_as_float(v);
}
#define NEG_INF_ENC 0x007FFFFFu   // fenc(-inf)

__global__ void fill_u32(unsigned* __restrict__ p, unsigned v, int n) {
  int i = blockIdx.x * blockDim.x + threadIdx.x;
  if (i < n) p[i] = v;
}

// ---------- precision staging: f32 -> bf16 ----------
__global__ void to_bf16_k(const float* __restrict__ a, unsigned short* __restrict__ o, int n) {
  int i = blockIdx.x * blockDim.x + threadIdx.x;
  if (i < n) o[i] = f2bf(a[i]);
}
// W[K,Nout] f32 -> Wt[Nout,K] bf16 (so B fragments are K-contiguous in memory)
__global__ void transpose_to_bf16(const float* __restrict__ W, unsigned short* __restrict__ Wt,
                                  int K, int Nout) {
  int i = blockIdx.x * blockDim.x + threadIdx.x;
  if (i >= K * Nout) return;
  int k = i / Nout, n = i % Nout;
  Wt[(size_t)n * K + k] = f2bf(W[i]);
}

// ---------- WMMA bf16 GEMM with async LDS staging ----------
//  C[M,Nout] = act(A[M,K] @ B[K,Nout] + bias), A bf16 row-major, Bt bf16 [Nout,K].
//  block = 128 threads (4 waves), tile 64(M) x 64(N), K-step 32.
//  Tiles staged with GLOBAL_LOAD_ASYNC_TO_LDS_B128 (ASYNCcnt protocol);
//  LDS rows padded to 48 bf16 (96B, 16B-aligned) for ds_load_b128 fragment reads.
#define LROW 48
__global__ __launch_bounds__(128) void gemm_bf16_wmma(
    const unsigned short* __restrict__ A, const unsigned short* __restrict__ Bt,
    const float* __restrict__ bias, float* __restrict__ C,
    int M, int K, int Nout, int relu)
{
  __shared__ __align__(16) unsigned short lA[64 * LROW];
  __shared__ __align__(16) unsigned short lB[64 * LROW];

  const int tid  = threadIdx.x;
  const int lane = tid & 31;
  const int wv   = tid >> 5;
  const int nl   = lane & 15;
  const int g    = lane >> 4;              // half-wave select
  const int row0 = blockIdx.y * 64;
  const int col0 = blockIdx.x * 64;

  // LDS byte offsets (generic LDS address low 32 bits == LDS offset)
  const unsigned aBase = (unsigned)(size_t)(void*)lA;
  const unsigned bBase = (unsigned)(size_t)(void*)lB;

  v8f acc[4];
  for (int t = 0; t < 4; ++t)
    for (int r = 0; r < 8; ++r) acc[t][r] = 0.0f;

  union FragU { int4 q[2]; v16bf v; };

  for (int kk = 0; kk < K; kk += 32) {
    // --- async-stage A and B tiles: 64 rows x 32 bf16 = 64B/row = 4x16B chunks ---
    #pragma unroll
    for (int it = 0; it < 2; ++it) {
      int id = tid + it * 128;             // 0..255 chunk id
      int r  = id >> 2;                    // tile row 0..63
      int c8 = (id & 3) * 8;               // bf16 element offset within row (0,8,16,24)
      int cb = c8 * 2;                     // byte offset (16B aligned)
      unsigned la = aBase + r * (LROW * 2) + cb;
      unsigned lb = bBase + r * (LROW * 2) + cb;
      const unsigned short* ga = A  + (size_t)(row0 + r) * K + kk + c8;
      const unsigned short* gb = Bt + (size_t)(col0 + r) * K + kk + c8;
      if (row0 + r < M)
        asm volatile("global_load_async_to_lds_b128 %0, %1, off"
                     :: "v"(la), "v"(ga) : "memory");
      if (col0 + r < Nout)
        asm volatile("global_load_async_to_lds_b128 %0, %1, off"
                     :: "v"(lb), "v"(gb) : "memory");
    }
    asm volatile("s_wait_asynccnt 0x0" ::: "memory");
    __syncthreads();

    // --- A fragment: row = wv*16 + nl ; elems[0..7]=K(g*8..), [8..15]=K(16+g*8..) ---
    FragU fa;
    const unsigned short* Ar = &lA[(wv * 16 + nl) * LROW + g * 8];
    fa.q[0] = *(const int4*)(Ar);
    fa.q[1] = *(const int4*)(Ar + 16);

    #pragma unroll
    for (int t = 0; t < 4; ++t) {
      FragU fb;   // col = t*16+nl ; elems[0..7]=K(g*8..), [8..15]=K(16+g*8..)
      const unsigned short* Br = &lB[(t * 16 + nl) * LROW + g * 8];
      fb.q[0] = *(const int4*)(Br);
      fb.q[1] = *(const int4*)(Br + 16);
      acc[t] = __builtin_amdgcn_wmma_f32_16x16x32_bf16(
          false, fa.v, false, fb.v, (short)0, acc[t], false, false);
    }
    __syncthreads();
  }

  // --- store: VGPR r -> M = r + g*8 ; lane -> N = nl ---
  for (int t = 0; t < 4; ++t) {
    int gc = col0 + t * 16 + nl;
    for (int r = 0; r < 8; ++r) {
      int gr = row0 + wv * 16 + g * 8 + r;
      if (gr < M && gc < Nout) {
        float v = acc[t][r] + (bias ? bias[gc] : 0.0f);
        if (relu) v = fmaxf(v, 0.0f);
        C[(size_t)gr * Nout + gc] = v;
      }
    }
  }
}

// ---------- attention logits: alsrc/aldst [N,H] = <h[n,hh,:], a[hh,:]> ----------
__global__ __launch_bounds__(256) void attn_logits(
    const float* __restrict__ h, const float* __restrict__ asrc,
    const float* __restrict__ adst, float* __restrict__ alsrc,
    float* __restrict__ aldst)
{
  int wid  = (blockIdx.x * blockDim.x + threadIdx.x) >> 5;
  int lane = threadIdx.x & 31;
  int n = wid >> 1, hh = wid & 1;
  if (n >= NN) return;
  const float* hp = h + (size_t)n * HE + hh * EMB;
  const float* as = asrc + hh * EMB;
  const float* ad = adst + hh * EMB;
  float s1 = 0.f, s2 = 0.f;
  for (int j = lane; j < EMB; j += 32) {
    float v = hp[j];
    s1 += v * as[j];
    s2 += v * ad[j];
  }
  for (int off = 16; off > 0; off >>= 1) {
    s1 += __shfl_down(s1, off, 32);
    s2 += __shfl_down(s2, off, 32);
  }
  if (lane == 0) { alsrc[n * NH + hh] = s1; aldst[n * NH + hh] = s2; }
}

// ---------- edge pass 1: e = leaky_relu(alsrc[src]+aldst[dst]); segment max ----------
__global__ void edge_max_k(const int* __restrict__ ei,
                           const float* __restrict__ alsrc,
                           const float* __restrict__ aldst,
                           float* __restrict__ e_ws,
                           unsigned* __restrict__ mvals)
{
  int i = blockIdx.x * blockDim.x + threadIdx.x;
  int eid = i >> 1, hh = i & 1;
  if (eid >= ETOT) return;
  int s = (eid < NE) ? ei[eid]      : (eid - NE);
  int d = (eid < NE) ? ei[NE + eid] : (eid - NE);
  float e = alsrc[s * NH + hh] + aldst[d * NH + hh];
  e = (e > 0.f) ? e : 0.2f * e;
  e_ws[(size_t)eid * NH + hh] = e;
  atomicMax(&mvals[d * NH + hh], fenc(e));
}

// ---------- edge pass 2: ex = exp(e - m[dst]); segment sum ----------
__global__ void edge_exp_k(const int* __restrict__ ei,
                           const unsigned* __restrict__ mvals,
                           float* __restrict__ e_ws,
                           float* __restrict__ ssum)
{
  int i = blockIdx.x * blockDim.x + threadIdx.x;
  int eid = i >> 1, hh = i & 1;
  if (eid >= ETOT) return;
  int d = (eid < NE) ? ei[NE + eid] : (eid - NE);
  float m  = fdec(mvals[d * NH + hh]);
  float ex = __expf(e_ws[(size_t)eid * NH + hh] - m);
  e_ws[(size_t)eid * NH + hh] = ex;
  atomicAdd(&ssum[d * NH + hh], ex);
}

// ---------- edge pass 3: agg[dst] += h[src] * alpha  (one edge per block) ----------
__global__ __launch_bounds__(128) void edge_aggregate(
    const int* __restrict__ ei, const float* __restrict__ h,
    const float* __restrict__ e_ws, const float* __restrict__ ssum,
    float* __restrict__ agg)
{
  int eid = blockIdx.x;
  int s = (eid < NE) ? ei[eid]      : (eid - NE);
  int d = (eid < NE) ? ei[NE + eid] : (eid - NE);
  int c  = threadIdx.x * 4;            // 0..508, covers H*EMB
  int hh = c >> 8;
  float alpha = e_ws[(size_t)eid * NH + hh] / ssum[d * NH + hh];
  float4 v = *(const float4*)(h + (size_t)s * HE + c);
  float* o = agg + (size_t)d * HE + c;
  atomicAdd(o + 0, v.x * alpha);
  atomicAdd(o + 1, v.y * alpha);
  atomicAdd(o + 2, v.z * alpha);
  atomicAdd(o + 3, v.w * alpha);
}

// ---------- bias + relu (in place), C = 512 ----------
__global__ void bias_relu_k(float* __restrict__ x, const float* __restrict__ b, int n) {
  int i = blockIdx.x * blockDim.x + threadIdx.x;
  if (i < n) x[i] = fmaxf(x[i] + b[i % HE], 0.0f);
}

// ---------- pooling ----------
__global__ void count_nodes(const int* __restrict__ batch, float* __restrict__ counts) {
  int i = blockIdx.x * blockDim.x + threadIdx.x;
  if (i < NN) atomicAdd(&counts[batch[i]], 1.0f);
}
__global__ void pool_accum(const float* __restrict__ x, const int* __restrict__ batch,
                           float* __restrict__ psum, unsigned* __restrict__ pmax) {
  int i = blockIdx.x * blockDim.x + threadIdx.x;
  if (i >= NN * EMB) return;
  int n = i >> 8, c = i & 255;
  int gg = batch[n];
  float v = x[(size_t)n * EMB + c];
  atomicAdd(&psum[gg * EMB + c], v);
  atomicMax(&pmax[gg * EMB + c], fenc(v));
}
__global__ void pool_finalize(const float* __restrict__ psum, const unsigned* __restrict__ pmax,
                              const float* __restrict__ counts, float* __restrict__ gacc) {
  int i = blockIdx.x * blockDim.x + threadIdx.x;
  if (i >= NG * EMB) return;
  int gg = i >> 8, c = i & 255;
  float cnt = counts[gg];
  float mean = psum[i] / fmaxf(cnt, 1.0f);
  float mx   = (cnt > 0.f) ? fdec(pmax[i]) : 0.0f;
  gacc[(size_t)gg * (2 * EMB) + c]       += mx;    // [max, mean] concat
  gacc[(size_t)gg * (2 * EMB) + EMB + c] += mean;
}

// ---------- final classifier: out[128,10] = gin[128,512] @ W[512,10] + b ----------
__global__ __launch_bounds__(256) void dense_out(
    const float* __restrict__ gin, const float* __restrict__ W,
    const float* __restrict__ b, float* __restrict__ out)
{
  int wid  = (blockIdx.x * blockDim.x + threadIdx.x) >> 5;
  int lane = threadIdx.x & 31;
  if (wid >= NG * NCLS) return;
  int gg = wid / NCLS, cls = wid % NCLS;
  const float* gp = gin + (size_t)gg * DENSE;
  float s = 0.f;
  for (int j = lane; j < DENSE; j += 32) s += gp[j] * W[j * NCLS + cls];
  for (int off = 16; off > 0; off >>= 1) s += __shfl_down(s, off, 32);
  if (lane == 0) out[gg * NCLS + cls] = s + b[cls];
}

// ================= host side =================
static inline dim3 g1(int n, int b = 256) { return dim3((n + b - 1) / b); }

extern "C" void kernel_launch(void* const* d_in, const int* in_sizes, int n_in,
                              void* d_out, int out_size, void* d_ws, size_t ws_size,
                              hipStream_t stream) {
  const float* x      = (const float*)d_in[0];
  const int*   ei     = (const int*)d_in[1];
  const int*   batch  = (const int*)d_in[2];
  const float* line1W = (const float*)d_in[21];
  const float* line1b = (const float*)d_in[22];
  const float* line2W = (const float*)d_in[23];
  const float* line2b = (const float*)d_in[24];

  // workspace carve-out
  size_t off = 0;
  auto carve = [&](size_t bytes) -> void* {
    void* p = (char*)d_ws + off;
    off += (bytes + 255) & ~(size_t)255;
    return p;
  };
  float*          h      = (float*)carve((size_t)NN * HE * 4);      // 40.96 MB
  float*          agg    = (float*)carve((size_t)NN * HE * 4);      // 40.96 MB
  float*          xcur   = (float*)carve((size_t)NN * EMB * 4);     // 20.48 MB
  unsigned short* abf    = (unsigned short*)carve((size_t)NN * HE * 2);  // 20.48 MB
  unsigned short* wtbf   = (unsigned short*)carve((size_t)DENSE * DENSE * 2);
  float*          alsrc  = (float*)carve((size_t)NN * NH * 4);
  float*          aldst  = (float*)carve((size_t)NN * NH * 4);
  unsigned*       mvals  = (unsigned*)carve((size_t)NN * NH * 4);
  float*          ssum   = (float*)carve((size_t)NN * NH * 4);
  float*          e_ws   = (float*)carve((size_t)ETOT * NH * 4);
  float*          psum   = (float*)carve((size_t)NG * EMB * 4);
  unsigned*       pmax   = (unsigned*)carve((size_t)NG * EMB * 4);
  float*          counts = (float*)carve((size_t)NG * 4);
  float*          gacc   = (float*)carve((size_t)NG * 2 * EMB * 4);
  float*          d1out  = (float*)carve((size_t)NG * DENSE * 4);

  // one-time zero/init
  fill_u32<<<g1(NG), 256, 0, stream>>>((unsigned*)counts, 0u, NG);
  fill_u32<<<g1(NG * 2 * EMB), 256, 0, stream>>>((unsigned*)gacc, 0u, NG * 2 * EMB);
  count_nodes<<<g1(NN), 256, 0, stream>>>(batch, counts);

  for (int L = 0; L < 3; ++L) {
    int base = 3 + L * 6;
    const float* attW  = (const float*)d_in[base + 0];
    const float* asrc  = (const float*)d_in[base + 1];
    const float* adst  = (const float*)d_in[base + 2];
    const float* attB  = (const float*)d_in[base + 3];
    const float* linW  = (const float*)d_in[base + 4];
    const float* linB  = (const float*)d_in[base + 5];
    const float* xin   = (L == 0) ? x : xcur;
    const int    K     = (L == 0) ? FIN : EMB;

    // h = xin @ attW   [NN,K] @ [K,512]   (bf16 staged, f32 accumulate)
    to_bf16_k<<<g1(NN * K), 256, 0, stream>>>(xin, abf, NN * K);
    transpose_to_bf16<<<g1(K * HE), 256, 0, stream>>>(attW, wtbf, K, HE);
    gemm_bf16_wmma<<<dim3(HE / 64, (NN + 63) / 64), 128, 0, stream>>>(
        abf, wtbf, nullptr, h, NN, K, HE, 0);

    attn_logits<<<g1(NN * NH * 32), 256, 0, stream>>>(h, asrc, adst, alsrc, aldst);

    fill_u32<<<g1(NN * NH), 256, 0, stream>>>(mvals, NEG_INF_ENC, NN * NH);
    fill_u32<<<g1(NN * NH), 256, 0, stream>>>((unsigned*)ssum, 0u, NN * NH);
    fill_u32<<<g1(NN * HE), 256, 0, stream>>>((unsigned*)agg, 0u, NN * HE);

    edge_max_k<<<g1(ETOT * NH), 256, 0, stream>>>(ei, alsrc, aldst, e_ws, mvals);
    edge_exp_k<<<g1(ETOT * NH), 256, 0, stream>>>(ei, mvals, e_ws, ssum);
    edge_aggregate<<<dim3(ETOT), 128, 0, stream>>>(ei, h, e_ws, ssum, agg);

    bias_relu_k<<<g1(NN * HE), 256, 0, stream>>>(agg, attB, NN * HE);

    // xcur = relu(agg @ linW + linB)   [NN,512] @ [512,256]
    to_bf16_k<<<g1(NN * HE), 256, 0, stream>>>(agg, abf, NN * HE);
    transpose_to_bf16<<<g1(HE * EMB), 256, 0, stream>>>(linW, wtbf, HE, EMB);
    gemm_bf16_wmma<<<dim3(EMB / 64, (NN + 63) / 64), 128, 0, stream>>>(
        abf, wtbf, linB, xcur, NN, HE, EMB, 1);

    fill_u32<<<g1(NG * EMB), 256, 0, stream>>>((unsigned*)psum, 0u, NG * EMB);
    fill_u32<<<g1(NG * EMB), 256, 0, stream>>>(pmax, NEG_INF_ENC, NG * EMB);
    pool_accum<<<g1(NN * EMB), 256, 0, stream>>>(xcur, batch, psum, pmax);
    pool_finalize<<<g1(NG * EMB), 256, 0, stream>>>(psum, pmax, counts, gacc);
  }

  // d1out = relu(gacc @ line1W + line1b)   [128,512] @ [512,512]
  to_bf16_k<<<g1(NG * 2 * EMB), 256, 0, stream>>>(gacc, abf, NG * 2 * EMB);
  transpose_to_bf16<<<g1(2 * EMB * DENSE), 256, 0, stream>>>(line1W, wtbf, 2 * EMB, DENSE);
  gemm_bf16_wmma<<<dim3(DENSE / 64, (NG + 63) / 64), 128, 0, stream>>>(
      abf, wtbf, line1b, d1out, NG, 2 * EMB, DENSE, 1);

  // out = d1out @ line2W + line2b   [128,512] @ [512,10]
  dense_out<<<dim3((NG * NCLS * 32 + 255) / 256), 256, 0, stream>>>(
      d1out, line2W, line2b, (float*)d_out);

  (void)in_sizes; (void)n_in; (void)out_size; (void)ws_size;
}
